// BaseModel_15788299780703
// MI455X (gfx1250) — matrix operand order repaired and checked by reading, not compiled, not run on hardware
//
#include <hip/hip_runtime.h>

// Problem constants (match reference)
#define DD       256
#define BG       128
#define NV_PER   20
#define NP_PER   500
#define NV_NODES (BG * NV_PER)   // 2560
#define NP_NODES (BG * NP_PER)   // 64000
#define EV       (NV_NODES * 8)  // 20480
#define EP       (NP_NODES * 8)  // 512000

typedef __attribute__((ext_vector_type(16))) __bf16 v16bf;
typedef __attribute__((ext_vector_type(8)))  float  v8f;

union Frag {
    v16bf v;
    unsigned int u[8];
    unsigned short h[16];
};
static_assert(sizeof(Frag) == 32, "frag size");

__device__ __forceinline__ unsigned short f2bf(float f) {
    unsigned int u = __float_as_uint(f);
    u += 0x7FFFu + ((u >> 16) & 1u);   // round-to-nearest-even
    return (unsigned short)(u >> 16);
}

// ---------------------------------------------------------------------------
// C[M,256] = A[M,256] @ W[256,256]   (bf16 WMMA, f32 accumulate)
// grid = (M/64, 4), block = 128 (4 waves). blockIdx.y = 64-column group.
// W slice (256 x 64) staged in LDS as bf16, pre-swizzled into B-fragment order.
// ---------------------------------------------------------------------------
__global__ __launch_bounds__(128)
void gemm_wmma(const float* __restrict__ A, const float* __restrict__ W,
               float* __restrict__ C, int M) {
    __shared__ unsigned int ldsW[8192];   // 32 KB
    const int tid = threadIdx.x;
    const int cg  = blockIdx.y;           // column group: cols [cg*64, cg*64+64)

    // Stage + swizzle W slice: fragment for (kblk, nblk, lane) is 8 dwords
    // holding bf16 pairs (K=kb+2j, K=kb+2j+1) for column n = cg*64+nblk*16+(lane&15),
    // kb = (j<4 ? 0 : 16) + (lane<16 ? 0 : 8) + kblk*32 - 2*(j&~3)   (matches A layout).
    for (int p = tid; p < 8192; p += 128) {
        int j    = p & 7;
        int lane = (p >> 3) & 31;
        int nblk = (p >> 8) & 3;
        int kblk = p >> 10;
        int col  = cg * 64 + nblk * 16 + (lane & 15);
        int k0   = kblk * 32 + ((j < 4) ? 0 : 16) + ((lane < 16) ? 0 : 8) + 2 * (j & 3);
        unsigned int lo = f2bf(W[(size_t)k0 * DD + col]);
        unsigned int hi = f2bf(W[(size_t)(k0 + 1) * DD + col]);
        ldsW[((kblk * 4 + nblk) * 32 + lane) * 8 + j] = lo | (hi << 16);
    }
    __syncthreads();

    const int wave = tid >> 5;
    const int lane = tid & 31;
    const int row0 = blockIdx.x * 64 + wave * 16;   // M % 64 == 0 always here
    const int kbA  = (lane < 16) ? 0 : 8;           // ISA 16-bit A layout
    const float* arow = A + (size_t)(row0 + (lane & 15)) * DD;

    v8f acc[4];
    #pragma unroll
    for (int n = 0; n < 4; ++n) acc[n] = (v8f)0.0f;

    for (int kblk = 0; kblk < 8; ++kblk) {
        Frag a;
        #pragma unroll
        for (int j = 0; j < 8; ++j) {
            int k0 = kblk * 32 + ((j < 4) ? 0 : 16) + kbA + 2 * (j & 3);
            float2 f = *(const float2*)(arow + k0);
            a.h[2 * j]     = f2bf(f.x);
            a.h[2 * j + 1] = f2bf(f.y);
        }
        #pragma unroll
        for (int nblk = 0; nblk < 4; ++nblk) {
            Frag b;
            const int base = ((kblk * 4 + nblk) * 32 + lane) * 8;
            #pragma unroll
            for (int j = 0; j < 8; ++j) b.u[j] = ldsW[base + j];
            acc[nblk] = __builtin_amdgcn_wmma_f32_16x16x32_bf16(
                false, a.v, false, b.v, (short)0, acc[nblk], false, false);
        }
    }

    // C/D layout: lanes 0-15 -> rows 0-7, lanes 16-31 -> rows 8-15; col = lane&15
    const int rbase = (lane >> 4) << 3;
    #pragma unroll
    for (int nblk = 0; nblk < 4; ++nblk) {
        int col = cg * 64 + nblk * 16 + (lane & 15);
        #pragma unroll
        for (int r = 0; r < 8; ++r)
            C[(size_t)(row0 + rbase + r) * DD + col] = acc[nblk][r];
    }
}

// ---------------------------------------------------------------------------
// Small helpers
// ---------------------------------------------------------------------------
__global__ void fillk(float* __restrict__ p, float v, long long n) {
    long long i = (long long)blockIdx.x * blockDim.x + threadIdx.x;
    if (i < n) p[i] = v;
}

// Y[M,256] = X[M,K] @ W[K,256] + b   (tiny K = 6 or 8)
__global__ void init_gemm(const float* __restrict__ X, const float* __restrict__ W,
                          const float* __restrict__ bias, float* __restrict__ Y,
                          int M, int K) {
    long long i = (long long)blockIdx.x * blockDim.x + threadIdx.x;
    if (i >= (long long)M * DD) return;
    int d = (int)(i & (DD - 1));
    int row = (int)(i >> 8);
    float s = bias[d];
    for (int k = 0; k < K; ++k) s += X[(size_t)row * K + k] * W[(size_t)k * DD + d];
    Y[i] = s;
}

__global__ void deg_count(const int* __restrict__ dst, float* __restrict__ deg, int E) {
    int e = blockIdx.x * blockDim.x + threadIdx.x;
    if (e < E) atomicAdd(&deg[dst[e]], 1.0f);
}

// agg[dst] += h[src] * rsqrt(deg[src]) * rsqrt(deg[dst]); one thread per (edge, dim)
__global__ void scatter_edges(const int* __restrict__ src, const int* __restrict__ dst,
                              const float* __restrict__ h, const float* __restrict__ deg,
                              float* __restrict__ agg, long long total) {
    long long i = (long long)blockIdx.x * blockDim.x + threadIdx.x;
    if (i >= total) return;
    int d = (int)(i & (DD - 1));
    int e = (int)(i >> 8);
    int s = src[e], t = dst[e];
    float coef = rsqrtf(deg[s]) * rsqrtf(deg[t]);
    atomicAdd(&agg[(size_t)t * DD + d], h[(size_t)s * DD + d] * coef);
}

// agg = (optional relu)(agg + hW/deg + bias)
__global__ void fuse_self(float* __restrict__ agg, const float* __restrict__ hW,
                          const float* __restrict__ deg, const float* __restrict__ bias,
                          long long total, int do_relu) {
    long long i = (long long)blockIdx.x * blockDim.x + threadIdx.x;
    if (i >= total) return;
    int d = (int)(i & (DD - 1));
    int row = (int)(i >> 8);
    float v = agg[i] + hW[i] / deg[row] + bias[d];
    if (do_relu) v = fmaxf(v, 0.0f);
    agg[i] = v;
}

__global__ void mean_pool(const float* __restrict__ node, float* __restrict__ graph,
                          int nper) {
    int i = blockIdx.x * blockDim.x + threadIdx.x;
    if (i >= BG * DD) return;
    int d = i & (DD - 1);
    int b = i >> 8;
    const float* p = node + (size_t)b * nper * DD + d;
    float s = 0.0f;
    for (int n = 0; n < nper; ++n) s += p[(size_t)n * DD];
    graph[i] = s * (1.0f / nper);
}

// comb[b,d] = p_graph + 2*v_graph + v_node[b,curr[b],d] + v_init[b,curr[b],d]
__global__ void make_comb(const float* __restrict__ pG, const float* __restrict__ vG,
                          const float* __restrict__ vNode, const float* __restrict__ vInit,
                          const int* __restrict__ curr, float* __restrict__ comb) {
    int i = blockIdx.x * blockDim.x + threadIdx.x;
    if (i >= BG * DD) return;
    int d = i & (DD - 1);
    int b = i >> 8;
    size_t idx = ((size_t)b * NV_PER + curr[b]) * DD + d;
    comb[i] = pG[i] + 2.0f * vG[i] + vNode[idx] + vInit[idx];
}

// out (currently p_node_emb) += p_init + comb[b,d]
__global__ void final_add(float* __restrict__ out, const float* __restrict__ pInit,
                          const float* __restrict__ comb) {
    long long i = (long long)blockIdx.x * blockDim.x + threadIdx.x;
    if (i >= (long long)NP_NODES * DD) return;
    int d = (int)(i & (DD - 1));
    int row = (int)(i >> 8);
    int b = row / NP_PER;
    out[i] += pInit[i] + comb[b * DD + d];
}

// ---------------------------------------------------------------------------
extern "C" void kernel_launch(void* const* d_in, const int* in_sizes, int n_in,
                              void* d_out, int out_size, void* d_ws, size_t ws_size,
                              hipStream_t stream) {
    const float* v_x      = (const float*)d_in[0];
    const float* p_x      = (const float*)d_in[1];
    const int*   v_src    = (const int*)d_in[2];
    const int*   v_dst    = (const int*)d_in[3];
    const int*   p_src    = (const int*)d_in[4];
    const int*   p_dst    = (const int*)d_in[5];
    const int*   curr     = (const int*)d_in[8];
    const float* v_init_W = (const float*)d_in[9];
    const float* v_init_b = (const float*)d_in[10];
    const float* v_W1     = (const float*)d_in[11];
    const float* v_b1     = (const float*)d_in[12];
    const float* v_W2     = (const float*)d_in[13];
    const float* v_b2     = (const float*)d_in[14];
    const float* p_init_W = (const float*)d_in[15];
    const float* p_init_b = (const float*)d_in[16];
    const float* p_W1     = (const float*)d_in[17];
    const float* p_b1     = (const float*)d_in[18];
    const float* p_W2     = (const float*)d_in[19];
    const float* p_b2     = (const float*)d_in[20];

    float* ws   = (float*)d_ws;
    float* P0   = ws; ws += (size_t)NP_NODES * DD;   // p init_emb
    float* P1   = ws; ws += (size_t)NP_NODES * DD;   // p GEMM out
    float* V0   = ws; ws += (size_t)NV_NODES * DD;   // v init_emb
    float* V1   = ws; ws += (size_t)NV_NODES * DD;   // v GEMM out
    float* V2   = ws; ws += (size_t)NV_NODES * DD;   // v agg / h / node_emb
    float* degP = ws; ws += NP_NODES;
    float* degV = ws; ws += NV_NODES;
    float* pG   = ws; ws += BG * DD;
    float* vG   = ws; ws += BG * DD;
    float* comb = ws; ws += BG * DD;
    float* P2   = (float*)d_out;                     // p agg / h / node_emb

    const long long vTot = (long long)NV_NODES * DD;
    const long long pTot = (long long)NP_NODES * DD;
    const int blk = 256;

    // ---------------- virtual net ----------------
    init_gemm<<<(int)((vTot + blk - 1) / blk), blk, 0, stream>>>(v_x, v_init_W, v_init_b, V0, NV_NODES, 6);
    fillk<<<(NV_NODES + blk - 1) / blk, blk, 0, stream>>>(degV, 1.0f, NV_NODES);
    deg_count<<<(EV + blk - 1) / blk, blk, 0, stream>>>(v_dst, degV, EV);

    gemm_wmma<<<dim3(NV_NODES / 64, 4), 128, 0, stream>>>(V0, v_W1, V1, NV_NODES);
    fillk<<<(int)((vTot + blk - 1) / blk), blk, 0, stream>>>(V2, 0.0f, vTot);
    scatter_edges<<<(int)(((long long)EV * DD + blk - 1) / blk), blk, 0, stream>>>(v_src, v_dst, V1, degV, V2, (long long)EV * DD);
    fuse_self<<<(int)((vTot + blk - 1) / blk), blk, 0, stream>>>(V2, V1, degV, v_b1, vTot, 1);

    gemm_wmma<<<dim3(NV_NODES / 64, 4), 128, 0, stream>>>(V2, v_W2, V1, NV_NODES);
    fillk<<<(int)((vTot + blk - 1) / blk), blk, 0, stream>>>(V2, 0.0f, vTot);
    scatter_edges<<<(int)(((long long)EV * DD + blk - 1) / blk), blk, 0, stream>>>(v_src, v_dst, V1, degV, V2, (long long)EV * DD);
    fuse_self<<<(int)((vTot + blk - 1) / blk), blk, 0, stream>>>(V2, V1, degV, v_b2, vTot, 0);

    mean_pool<<<(BG * DD + blk - 1) / blk, blk, 0, stream>>>(V2, vG, NV_PER);

    // ---------------- physical net ----------------
    init_gemm<<<(int)((pTot + blk - 1) / blk), blk, 0, stream>>>(p_x, p_init_W, p_init_b, P0, NP_NODES, 8);
    fillk<<<(NP_NODES + blk - 1) / blk, blk, 0, stream>>>(degP, 1.0f, NP_NODES);
    deg_count<<<(EP + blk - 1) / blk, blk, 0, stream>>>(p_dst, degP, EP);

    gemm_wmma<<<dim3(NP_NODES / 64, 4), 128, 0, stream>>>(P0, p_W1, P1, NP_NODES);
    fillk<<<(int)((pTot + blk - 1) / blk), blk, 0, stream>>>(P2, 0.0f, pTot);
    scatter_edges<<<(int)(((long long)EP * DD + blk - 1) / blk), blk, 0, stream>>>(p_src, p_dst, P1, degP, P2, (long long)EP * DD);
    fuse_self<<<(int)((pTot + blk - 1) / blk), blk, 0, stream>>>(P2, P1, degP, p_b1, pTot, 1);

    gemm_wmma<<<dim3(NP_NODES / 64, 4), 128, 0, stream>>>(P2, p_W2, P1, NP_NODES);
    fillk<<<(int)((pTot + blk - 1) / blk), blk, 0, stream>>>(P2, 0.0f, pTot);
    scatter_edges<<<(int)(((long long)EP * DD + blk - 1) / blk), blk, 0, stream>>>(p_src, p_dst, P1, degP, P2, (long long)EP * DD);
    fuse_self<<<(int)((pTot + blk - 1) / blk), blk, 0, stream>>>(P2, P1, degP, p_b2, pTot, 0);

    mean_pool<<<(BG * DD + blk - 1) / blk, blk, 0, stream>>>(P2, pG, NP_PER);

    // ---------------- fusion ----------------
    make_comb<<<(BG * DD + blk - 1) / blk, blk, 0, stream>>>(pG, vG, V2, V0, curr, comb);
    final_add<<<(int)((pTot + blk - 1) / blk), blk, 0, stream>>>(P2, P0, comb);
}
